// Transcoders_86294482911894
// MI455X (gfx1250) — matrix operand order repaired and checked by reading, not compiled, not run on hardware
//
#include <hip/hip_runtime.h>
#include <stdint.h>

typedef __bf16 bf16;
typedef bf16  v16bf __attribute__((ext_vector_type(16)));
typedef bf16  v8bf  __attribute__((ext_vector_type(8)));
typedef float v8f   __attribute__((ext_vector_type(8)));

#define B_ROWS 8192
#define DDIM   768
#define LDIM   12288
#define KTOP1  32
#define KTOP2  128
#define EPT    48   // LDIM / 256 elements per thread in topk

// ---------------- conversion / stats kernels ----------------

__global__ void cvt_bf16_kernel(const float* __restrict__ src, bf16* __restrict__ dst, int n) {
  int i = blockIdx.x * blockDim.x + threadIdx.x;
  int stride = gridDim.x * blockDim.x;
  for (; i < n; i += stride) dst[i] = (bf16)src[i];
}

// W_skip is [K, N]; we need B-operand rows indexed by N with K contiguous -> transpose.
__global__ void cvtT_kernel(const float* __restrict__ src, bf16* __restrict__ dst) {
  int i = blockIdx.x * blockDim.x + threadIdx.x;
  if (i < DDIM * DDIM) {
    int k = i / DDIM, d = i % DDIM;
    dst[(size_t)d * DDIM + k] = (bf16)src[i];
  }
}

__global__ void zero_kernel(float* colsum, float* colsumsq, double* acc) {
  int t = blockIdx.x * blockDim.x + threadIdx.x;
  if (t < DDIM) { colsum[t] = 0.f; colsumsq[t] = 0.f; }
  if (t < 8) acc[t] = 0.0;
}

// per-column sum & sum-of-squares of x for total_variance
__global__ __launch_bounds__(256)
void colstats_kernel(const float* __restrict__ x, float* colsum, float* colsumsq) {
  int t = threadIdx.x;
  int r0 = blockIdx.x * 256;
  float s0=0,s1=0,s2=0,q0=0,q1=0,q2=0;
  for (int r = 0; r < 256; ++r) {
    const float* xr = x + (size_t)(r0 + r) * DDIM;
    float v0 = xr[t], v1 = xr[t+256], v2 = xr[t+512];
    s0+=v0; q0+=v0*v0; s1+=v1; q1+=v1*v1; s2+=v2; q2+=v2*v2;
  }
  atomicAdd(&colsum[t],     s0); atomicAdd(&colsumsq[t],     q0);
  atomicAdd(&colsum[t+256], s1); atomicAdd(&colsumsq[t+256], q1);
  atomicAdd(&colsum[t+512], s2); atomicAdd(&colsumsq[t+512], q2);
}

// ---------------- bf16 WMMA GEMM:  C[m,n] = sum_k A[m,k]*W[n,k] + bias[n] ----------------
// A: [M,K] bf16 row-major; W: [N,K] bf16 row-major; C: [M,N] f32.
// Block tile 128x128, BK=32, 128 threads (4 waves in 2x2), wave tile 64x64 (4x4 WMMA).
// Double-buffered LDS: 1 barrier per k-step; next tile's global loads issued pre-barrier.

#define LDT 48  // LDS row stride in bf16 elems: 96B -> 16/32B aligned fragments, spread banks

__global__ __launch_bounds__(128)
void gemm_bf16_kernel(const bf16* __restrict__ A, const bf16* __restrict__ W,
                      const float* __restrict__ bias, float* __restrict__ C,
                      int M, int N, int K) {
  __shared__ __align__(32) bf16 As[2][128 * LDT];
  __shared__ __align__(32) bf16 Bs[2][128 * LDT];
  int tid  = threadIdx.x;
  int lane = tid & 31, wv = tid >> 5;
  int waveM = (wv >> 1) * 64, waveN = (wv & 1) * 64;
  int rl = lane & 15, half = lane >> 4;
  int blockM = blockIdx.y * 128, blockN = blockIdx.x * 128;

  // staging: 128 rows x 32 k bf16 per matrix = 512 chunks of 16B; 4 chunks/thread/matrix
  int r0  = tid >> 2;            // 0..31
  int off = (tid & 3) * 8;       // 0,8,16,24
  const bf16* gA[4]; const bf16* gB[4]; int ls[4];
  for (int i = 0; i < 4; ++i) {
    int r = r0 + 32 * i;
    gA[i] = A + (size_t)(blockM + r) * K + off;
    gB[i] = W + (size_t)(blockN + r) * K + off;
    ls[i] = r * LDT + off;
  }

  v8f acc[4][4] = {};

  // preload tile 0 into buffer 0
  v8bf ra[4], rb[4];
  for (int i = 0; i < 4; ++i) { ra[i] = *(const v8bf*)gA[i]; rb[i] = *(const v8bf*)gB[i]; }
  for (int i = 0; i < 4; ++i) { *(v8bf*)&As[0][ls[i]] = ra[i]; *(v8bf*)&Bs[0][ls[i]] = rb[i]; }

  int cur = 0;
  for (int k0 = 0; k0 < K; k0 += 32) {
    bool more = (k0 + 32) < K;
    if (more) {  // issue next tile's global loads before syncing (overlap with WMMA)
      for (int i = 0; i < 4; ++i) {
        gA[i] += 32; gB[i] += 32;
        ra[i] = *(const v8bf*)gA[i]; rb[i] = *(const v8bf*)gB[i];
      }
    }
    __syncthreads();  // buffer `cur` fully written (stores from previous iteration)

    // A fragment (16x32): lane<16 holds row rl, K {0..7,16..23}; lane>=16: {8..15,24..31}
    v16bf afrag[4], bfrag[4];
    for (int mi = 0; mi < 4; ++mi) {
      int rr = (waveM + mi*16 + rl) * LDT;
      v8bf* ah = (v8bf*)&afrag[mi];
      ah[0] = *(const v8bf*)&As[cur][rr + half*8];
      ah[1] = *(const v8bf*)&As[cur][rr + 16 + half*8];
    }
    // B fragment (32x16): lane<16 = col rl K {0..7,16..23}; lane>=16 = col rl K {8..15,24..31}
    for (int ni = 0; ni < 4; ++ni) {
      int rr = (waveN + ni*16 + rl) * LDT;
      v8bf* bh = (v8bf*)&bfrag[ni];
      bh[0] = *(const v8bf*)&Bs[cur][rr + half*8];
      bh[1] = *(const v8bf*)&Bs[cur][rr + 16 + half*8];
    }
    for (int mi = 0; mi < 4; ++mi)
      for (int ni = 0; ni < 4; ++ni)
        acc[mi][ni] = __builtin_amdgcn_wmma_f32_16x16x32_bf16(
            false, afrag[mi], false, bfrag[ni], (short)0, acc[mi][ni], false, false);

    if (more) {  // write next tile into the other buffer (no conflict with readers of `cur`)
      int nxt = cur ^ 1;
      for (int i = 0; i < 4; ++i) {
        *(v8bf*)&As[nxt][ls[i]] = ra[i];
        *(v8bf*)&Bs[nxt][ls[i]] = rb[i];
      }
      cur = nxt;
    }
  }

  // C/D layout: lanes 0-15 -> N=lane, M=j ; lanes 16-31 -> N=lane-16, M=j+8
  for (int mi = 0; mi < 4; ++mi)
    for (int ni = 0; ni < 4; ++ni) {
      int col = blockN + waveN + ni*16 + rl;
      float bv = bias[col];
      for (int j = 0; j < 8; ++j) {
        int row = blockM + waveM + mi*16 + half*8 + j;
        C[(size_t)row * N + col] = acc[mi][ni][j] + bv;
      }
    }
}

// ---------------- top-128 per row: register-resident radix threshold + rank sort ----------------

__global__ __launch_bounds__(256)
void topk_kernel(const float* __restrict__ pre, float* __restrict__ topv,
                 int* __restrict__ topi, double* acc) {
  __shared__ uint32_t ckey[160];
  __shared__ int      cidx[160];
  __shared__ float    sval[KTOP2];
  __shared__ int      sidx[KTOP2];
  __shared__ float    red[256];
  __shared__ unsigned scount;
  __shared__ int      cnt;

  int row = blockIdx.x, t = threadIdx.x;
  const float* p = pre + (size_t)row * LDIM;

  // order-preserving float->uint keys, kept in registers
  uint32_t kreg[EPT];
#pragma unroll
  for (int j = 0; j < EPT; ++j) {
    uint32_t b = __float_as_uint(p[t + j * 256]);
    kreg[j] = (b & 0x80000000u) ? ~b : (b | 0x80000000u);
  }

  // build largest threshold thr with count(key >= thr) >= 128, MSB-first
  uint32_t thr = 0;
  for (int bit = 31; bit >= 0; --bit) {
    uint32_t cand = thr | (1u << bit);
    if (t == 0) scount = 0;
    __syncthreads();
    unsigned local = 0;
#pragma unroll
    for (int j = 0; j < EPT; ++j) local += (kreg[j] >= cand) ? 1u : 0u;
    for (int o = 16; o > 0; o >>= 1) local += __shfl_down(local, o, 32);
    if ((t & 31) == 0) atomicAdd(&scount, local);
    __syncthreads();
    if (scount >= KTOP2) thr = cand;
    __syncthreads();
  }

  // compact survivors (indices reconstructed from lane/chunk position)
  if (t == 0) cnt = 0;
  __syncthreads();
#pragma unroll
  for (int j = 0; j < EPT; ++j) {
    if (kreg[j] >= thr) {
      int pos = atomicAdd(&cnt, 1);
      if (pos < KTOP2) { ckey[pos] = kreg[j]; cidx[pos] = t + j * 256; }
    }
  }
  __syncthreads();

  // O(n^2) rank sort of 128 survivors (descending)
  if (t < KTOP2) {
    uint32_t k = ckey[t];
    int r = 0;
    for (int j = 0; j < KTOP2; ++j) {
      uint32_t kj = ckey[j];
      r += (kj > k || (kj == k && j < t)) ? 1 : 0;
    }
    uint32_t b = (k & 0x80000000u) ? (k & 0x7FFFFFFFu) : ~k;
    sval[r] = __uint_as_float(b);
    sidx[r] = cidx[t];
  }
  __syncthreads();

  // store relu'd values + indices; accumulate encoded-sum (top32 counts twice)
  float contrib = 0.f;
  if (t < KTOP2) {
    float v = fmaxf(sval[t], 0.f);
    topv[row * KTOP2 + t] = v;
    topi[row * KTOP2 + t] = sidx[t];
    contrib = v * ((t < KTOP1) ? 2.f : 1.f);
  }
  red[t] = contrib;
  __syncthreads();
  for (int s = 128; s > 0; s >>= 1) { if (t < s) red[t] += red[t + s]; __syncthreads(); }
  if (t == 0) atomicAdd(&acc[0], (double)red[0]);
}

// ---------------- sparse decode + loss reduction ----------------

__global__ __launch_bounds__(256)
void decode_kernel(const float* __restrict__ skip, const float* __restrict__ topv,
                   const int* __restrict__ topi, const float* __restrict__ Wdec,
                   const float* __restrict__ mlp, double* acc) {
  __shared__ float lv[KTOP2];
  __shared__ int   li[KTOP2];
  __shared__ float r1[256], r2[256], r3[256];
  int row = blockIdx.x, t = threadIdx.x;
  if (t < KTOP2) { lv[t] = topv[row*KTOP2 + t]; li[t] = topi[row*KTOP2 + t]; }
  __syncthreads();

  size_t ro = (size_t)row * DDIM;
  float a0 = skip[ro + t], a1 = skip[ro + t + 256], a2 = skip[ro + t + 512];
  float d0 = 0.f, d1 = 0.f, d2 = 0.f;
  for (int j = 0; j < KTOP2; ++j) {
    if (j == KTOP1) { d0 = a0; d1 = a1; d2 = a2; }   // snapshot = decoded (top-32)
    float v = lv[j];
    const float* wr = Wdec + (size_t)li[j] * DDIM;
    a0 += v * wr[t]; a1 += v * wr[t + 256]; a2 += v * wr[t + 512];
  }
  float m0 = mlp[ro + t], m1 = mlp[ro + t + 256], m2 = mlp[ro + t + 512];
  float e, sse1 = 0.f, sse2 = 0.f;
  e = m0 - d0; sse1 += e*e; e = m1 - d1; sse1 += e*e; e = m2 - d2; sse1 += e*e;
  e = m0 - a0; sse2 += e*e; e = m1 - a1; sse2 += e*e; e = m2 - a2; sse2 += e*e;
  float ds = d0 + d1 + d2 + a0 + a1 + a2;

  r1[t] = sse1; r2[t] = sse2; r3[t] = ds;
  __syncthreads();
  for (int s = 128; s > 0; s >>= 1) {
    if (t < s) { r1[t] += r1[t+s]; r2[t] += r2[t+s]; r3[t] += r3[t+s]; }
    __syncthreads();
  }
  if (t == 0) {
    atomicAdd(&acc[1], (double)r1[0]);
    atomicAdd(&acc[2], (double)r2[0]);
    atomicAdd(&acc[3], (double)r3[0]);
  }
}

__global__ __launch_bounds__(256)
void finalize_kernel(const float* __restrict__ colsum, const float* __restrict__ colsumsq,
                     const double* __restrict__ acc, float* __restrict__ out) {
  __shared__ double red[256];
  int t = threadIdx.x;
  double tv = 0.0;
  for (int c = t; c < DDIM; c += 256) {
    double s = colsum[c], q = colsumsq[c];
    tv += q - s * s / (double)B_ROWS;
  }
  red[t] = tv;
  __syncthreads();
  for (int s = 128; s > 0; s >>= 1) { if (t < s) red[t] += red[t+s]; __syncthreads(); }
  if (t == 0) {
    double TV = red[0];
    out[0] = (float)(acc[0] / ((double)B_ROWS * (double)LDIM));
    out[1] = (float)(acc[3] / ((double)B_ROWS * (double)DDIM));
    out[2] = (float)(acc[1] / TV + acc[2] / TV / 8.0);
  }
}

// ---------------- launch ----------------

extern "C" void kernel_launch(void* const* d_in, const int* in_sizes, int n_in,
                              void* d_out, int out_size, void* d_ws, size_t ws_size,
                              hipStream_t stream) {
  const float* x     = (const float*)d_in[0];
  const float* mlp   = (const float*)d_in[1];
  const float* Wenc  = (const float*)d_in[2];
  const float* benc  = (const float*)d_in[3];
  const float* Wdec  = (const float*)d_in[4];
  const float* bdec  = (const float*)d_in[5];
  const float* Wskip = (const float*)d_in[6];
  float* out = (float*)d_out;

  char* w = (char*)d_ws;
  size_t o = 0;
  float* pre     = (float*)(w + o); o += (size_t)B_ROWS * LDIM * 4;   // 402.7 MB
  bf16*  xb      = (bf16*) (w + o); o += (size_t)B_ROWS * DDIM * 2;
  bf16*  wencb   = (bf16*) (w + o); o += (size_t)LDIM   * DDIM * 2;
  bf16*  wskipTb = (bf16*) (w + o); o += (size_t)DDIM   * DDIM * 2;
  float* skip    = (float*)(w + o); o += (size_t)B_ROWS * DDIM * 4;
  float* topv    = (float*)(w + o); o += (size_t)B_ROWS * KTOP2 * 4;
  int*   topi    = (int*)  (w + o); o += (size_t)B_ROWS * KTOP2 * 4;
  float* colsum  = (float*)(w + o); o += (size_t)DDIM * 4;
  float* colsq   = (float*)(w + o); o += (size_t)DDIM * 4;
  double* acc    = (double*)(w + o);

  zero_kernel<<<4, 256, 0, stream>>>(colsum, colsq, acc);
  cvt_bf16_kernel<<<4096, 256, 0, stream>>>(x, xb, B_ROWS * DDIM);
  cvt_bf16_kernel<<<4096, 256, 0, stream>>>(Wenc, wencb, LDIM * DDIM);
  cvtT_kernel<<<(DDIM*DDIM + 255)/256, 256, 0, stream>>>(Wskip, wskipTb);
  colstats_kernel<<<B_ROWS/256, 256, 0, stream>>>(x, colsum, colsq);

  gemm_bf16_kernel<<<dim3(LDIM/128, B_ROWS/128), 128, 0, stream>>>(
      xb, wencb, benc, pre, B_ROWS, LDIM, DDIM);
  gemm_bf16_kernel<<<dim3(DDIM/128, B_ROWS/128), 128, 0, stream>>>(
      xb, wskipTb, bdec, skip, B_ROWS, DDIM, DDIM);

  topk_kernel<<<B_ROWS, 256, 0, stream>>>(pre, topv, topi, acc);
  decode_kernel<<<B_ROWS, 256, 0, stream>>>(skip, topv, topi, Wdec, mlp, acc);
  finalize_kernel<<<1, 256, 0, stream>>>(colsum, colsq, acc, out);
}
